// EGNN_11630771437666
// MI455X (gfx1250) — matrix operand order repaired and checked by reading, not compiled, not run on hardware
//
#include <hip/hip_runtime.h>
#include <math.h>

// ---------------------------------------------------------------------------
// EGNN forward for MI455X (gfx1250), f16 WMMA with f32 accumulate.
//   N=50000 nodes, E=800000 edges, H=96, 4 layers, 256 graphs.
// GEMMs are [M,K]x[K,96], K=96 (embed stage 1: K=64) -> 6 N-tiles x K/32 k-steps of v_wmma_f32_16x16x32_f16.
// ---------------------------------------------------------------------------

#define N_NODES 50000
#define N_EDGES 800000
#define HDIM    96
#define NGRAPH  256
#define NLAYER  4

typedef __attribute__((ext_vector_type(16))) _Float16 v16h;
typedef __attribute__((ext_vector_type(8)))  _Float16 v8h;
typedef __attribute__((ext_vector_type(8)))  float    v8f;

union FragU { v16h v; v8h h[2]; };

// A-matrix 16x32 f16 fragment: lane(0-15)=row, lanes 16-31 same rows, different K half.
// elements 0..7  -> k = base + hi*8 + 0..7
// elements 8..15 -> k = base + hi*8 + 16..23
__device__ inline v16h load_a_frag(const _Float16* p) {
  FragU f;
  f.h[0] = *(const v8h*)(p);
  f.h[1] = *(const v8h*)(p + 16);
  return f.v;
}

__device__ inline float silu_f(float v) { return v * (1.0f / (1.0f + __expf(-v))); }

// ---------------------------------------------------------------------------
// Weight packing: fp32 [K,96] row-major -> B-fragment order for 16x16x32 f16 WMMA.
// Packed half index: ((s*6 + nt)*32 + lane)*16 + e
//   k = s*32 + (lane>>4)*8 + (e&7) + (e>>3)*16 ; n = nt*16 + (lane&15)
// K padded with zeros up to ksteps*32.
// ---------------------------------------------------------------------------
__global__ void pack_w_kernel(const float* __restrict__ src, _Float16* __restrict__ dst,
                              int Kact, int ksteps) {
  int t = blockIdx.x * blockDim.x + threadIdx.x;
  int total = ksteps * 6 * 32 * 16;
  if (t >= total) return;
  int e    = t & 15;
  int lane = (t >> 4) & 31;
  int rem  = t >> 9;           // s*6 + nt
  int nt   = rem % 6;
  int s    = rem / 6;
  int hl = lane & 15, hi = lane >> 4;
  int k = s * 32 + hi * 8 + (e & 7) + (e >> 3) * 16;
  int n = nt * 16 + hl;
  dst[t] = (k < Kact) ? (_Float16)src[k * HDIM + n] : (_Float16)0.0f;
}

// Build [N,64] f16 input: x(11) || pe(24) || zero pad
__global__ void build_hin_kernel(const float* __restrict__ x, const float* __restrict__ pe,
                                 _Float16* __restrict__ hin, int n) {
  int t = blockIdx.x * blockDim.x + threadIdx.x;
  if (t >= n * 64) return;
  int row = t >> 6, j = t & 63;
  float v = 0.0f;
  if (j < 11)      v = x[row * 11 + j];
  else if (j < 35) v = pe[row * 24 + (j - 11)];
  hin[t] = (_Float16)v;
}

__global__ void f32_to_f16_kernel(const float* __restrict__ in, _Float16* __restrict__ out, int n) {
  int i = blockIdx.x * blockDim.x + threadIdx.x;
  if (i < n) out[i] = (_Float16)in[i];
}

__global__ void dist_kernel(const float* __restrict__ pos, const int* __restrict__ ei,
                            float* __restrict__ dist, int e) {
  int i = blockIdx.x * blockDim.x + threadIdx.x;
  if (i >= e) return;
  int s = ei[i], r = ei[e + i];
  float dx = pos[s * 3 + 0] - pos[r * 3 + 0];
  float dy = pos[s * 3 + 1] - pos[r * 3 + 1];
  float dz = pos[s * 3 + 2] - pos[r * 3 + 2];
  dist[i] = sqrtf(dx * dx + dy * dy + dz * dz);
}

// ---------------------------------------------------------------------------
// Fused 2-layer MLP over row tiles of 16, 4 waves (64 rows) per block.
//   stage1: acc = A1[idx1]·W1a (+ A2[idx2]·W1b) + bias1 (+ r1val·r1w), SiLU
//           -> per-wave 16x96 f16 tile in LDS (operand-layout transpose)
//   stage2: out = LDS·W2 + bias2 (opt SiLU) (opt +residual)
//           -> outF(f32) / outH(f16) / atomicAdd scatter (segment sum)
// Row indices are clamped so EXEC stays all-ones through every WMMA.
// ---------------------------------------------------------------------------
__global__ __launch_bounds__(128) void fused_mlp2_kernel(
    const _Float16* __restrict__ A1, int lda1, int ks1, const int* __restrict__ idx1,
    const _Float16* __restrict__ A2, const int* __restrict__ idx2,
    const _Float16* __restrict__ W1a, const _Float16* __restrict__ W1b,
    const float* __restrict__ bias1,
    const float* __restrict__ r1val, const float* __restrict__ r1w,
    const _Float16* __restrict__ W2, const float* __restrict__ bias2,
    int do_silu2,
    const float* __restrict__ residual,
    float* __restrict__ outF, _Float16* __restrict__ outH,
    float* __restrict__ outAtomic, const int* __restrict__ scatterIdx,
    int M) {
  __shared__ __align__(32) _Float16 lds[4 * 16 * 96];
  const int tid  = threadIdx.x;
  const int wave = tid >> 5;
  const int lane = tid & 31;
  const int hl   = lane & 15;
  const int hi   = lane >> 4;
  const int mBase = blockIdx.x * 64 + wave * 16;

  int arow = mBase + hl;
  if (arow > M - 1) arow = M - 1;
  const int g1 = idx1 ? idx1[arow] : arow;
  const _Float16* a1p = A1 + (size_t)g1 * lda1 + hi * 8;

  v8f acc[6] = {};
  for (int s = 0; s < ks1; ++s) {
    v16h a = load_a_frag(a1p + s * 32);
#pragma unroll
    for (int nt = 0; nt < 6; ++nt) {
      v16h b = *(const v16h*)(W1a + ((size_t)(s * 6 + nt) * 32 + lane) * 16);
      acc[nt] = __builtin_amdgcn_wmma_f32_16x16x32_f16(false, a, false, b,
                                                       (short)0, acc[nt], false, false);
    }
  }
  if (A2) {
    const int g2 = idx2 ? idx2[arow] : arow;
    const _Float16* a2p = A2 + (size_t)g2 * HDIM + hi * 8;
    for (int s = 0; s < 3; ++s) {
      v16h a = load_a_frag(a2p + s * 32);
#pragma unroll
      for (int nt = 0; nt < 6; ++nt) {
        v16h b = *(const v16h*)(W1b + ((size_t)(s * 6 + nt) * 32 + lane) * 16);
        acc[nt] = __builtin_amdgcn_wmma_f32_16x16x32_f16(false, a, false, b,
                                                         (short)0, acc[nt], false, false);
      }
    }
  }

  // stage-1 epilogue -> LDS f16 tile (C layout: col = nt*16+hl, row = hi*8+r)
  _Float16* lbase = &lds[wave * 16 * 96];
#pragma unroll
  for (int nt = 0; nt < 6; ++nt) {
    const int col = nt * 16 + hl;
    const float bv = bias1[col];
    const float rw = r1w ? r1w[col] : 0.0f;
#pragma unroll
    for (int r = 0; r < 8; ++r) {
      const int lrow = hi * 8 + r;
      float v = acc[nt][r] + bv;
      if (r1val) {
        int grow = mBase + lrow;
        if (grow > M - 1) grow = M - 1;
        v += r1val[grow] * rw;
      }
      v = silu_f(v);
      lbase[lrow * 96 + col] = (_Float16)v;
    }
  }
  __syncthreads();

  // stage 2: K=96 from LDS
  v8f acc2[6] = {};
  const _Float16* lp = &lds[wave * 16 * 96 + hl * 96 + hi * 8];
#pragma unroll
  for (int s = 0; s < 3; ++s) {
    FragU f;
    f.h[0] = *(const v8h*)(lp + s * 32);
    f.h[1] = *(const v8h*)(lp + s * 32 + 16);
    v16h a = f.v;
#pragma unroll
    for (int nt = 0; nt < 6; ++nt) {
      v16h b = *(const v16h*)(W2 + ((size_t)(s * 6 + nt) * 32 + lane) * 16);
      acc2[nt] = __builtin_amdgcn_wmma_f32_16x16x32_f16(false, a, false, b,
                                                        (short)0, acc2[nt], false, false);
    }
  }

  // stage-2 epilogue (divergence only here, after all WMMA)
#pragma unroll
  for (int nt = 0; nt < 6; ++nt) {
    const int col = nt * 16 + hl;
    const float bv = bias2[col];
#pragma unroll
    for (int r = 0; r < 8; ++r) {
      const int grow = mBase + hi * 8 + r;
      float v = acc2[nt][r] + bv;
      if (do_silu2) v = silu_f(v);
      if (grow < M) {
        const size_t o = (size_t)grow * HDIM + col;
        if (residual) v += residual[o];
        if (outF) outF[o] = v;
        if (outH) outH[o] = (_Float16)v;
        if (outAtomic) atomicAdd(&outAtomic[(size_t)scatterIdx[grow] * HDIM + col], v);
      }
    }
  }
}

// Readout: out[g] = silu(pooled[g]·ro_w1 + b1) · ro_w2 + b2   (96 threads / graph)
__global__ void readout_kernel(const float* __restrict__ pooled,
                               const float* __restrict__ w1, const float* __restrict__ b1,
                               const float* __restrict__ w2, const float* __restrict__ b2,
                               float* __restrict__ out) {
  int g = blockIdx.x, j = threadIdx.x;
  __shared__ float sh[96];
  const float* p = pooled + g * HDIM;
  float acc = b1[j];
  for (int k = 0; k < HDIM; ++k) acc += p[k] * w1[k * HDIM + j];
  sh[j] = silu_f(acc) * w2[j];
  __syncthreads();
  if (j == 0) {
    float s = b2[0];
    for (int k = 0; k < HDIM; ++k) s += sh[k];
    out[g] = s;
  }
}

// ---------------------------------------------------------------------------
extern "C" void kernel_launch(void* const* d_in, const int* in_sizes, int n_in,
                              void* d_out, int out_size, void* d_ws, size_t ws_size,
                              hipStream_t stream) {
  const float* x        = (const float*)d_in[0];
  const float* pos      = (const float*)d_in[1];
  const float* pe       = (const float*)d_in[2];
  const int*   ei       = (const int*)d_in[3];   // [2,E]: send row0, rec row1
  const int*   batch    = (const int*)d_in[4];
  const float* embed_w1 = (const float*)d_in[5];
  const float* embed_b1 = (const float*)d_in[6];
  const float* embed_w2 = (const float*)d_in[7];
  const float* embed_b2 = (const float*)d_in[8];
  const float* msg_w1   = (const float*)d_in[9];   // [4,193,96]
  const float* msg_b1   = (const float*)d_in[10];  // [4,96]
  const float* msg_w2   = (const float*)d_in[11];  // [4,96,96]
  const float* msg_b2   = (const float*)d_in[12];
  const float* upd_w1   = (const float*)d_in[13];  // [4,192,96]
  const float* upd_b1   = (const float*)d_in[14];
  const float* upd_w2   = (const float*)d_in[15];
  const float* upd_b2   = (const float*)d_in[16];
  const float* pre_w1   = (const float*)d_in[17];
  const float* pre_b1   = (const float*)d_in[18];
  const float* pre_w2   = (const float*)d_in[19];
  const float* pre_b2   = (const float*)d_in[20];
  const float* ro_w1    = (const float*)d_in[21];
  const float* ro_b1    = (const float*)d_in[22];
  const float* ro_w2    = (const float*)d_in[23];  // [96,1]
  const float* ro_b2    = (const float*)d_in[24];

  const int N = N_NODES, E = N_EDGES;
  const int* send = ei;
  const int* rec  = ei + E;

  // workspace partition (256B aligned)
  char* ws = (char*)d_ws;
  size_t off = 0;
  auto alloc = [&](size_t bytes) -> char* {
    char* p = ws + off;
    off += (bytes + 255) & ~(size_t)255;
    return p;
  };
  float*    hA     = (float*)alloc((size_t)N * 96 * 4);
  float*    hB     = (float*)alloc((size_t)N * 96 * 4);
  _Float16* h16A   = (_Float16*)alloc((size_t)N * 96 * 2);
  _Float16* h16B   = (_Float16*)alloc((size_t)N * 96 * 2);
  float*    aggr   = (float*)alloc((size_t)N * 96 * 4);
  _Float16* aggr16 = (_Float16*)alloc((size_t)N * 96 * 2);
  _Float16* hin16  = (_Float16*)alloc((size_t)N * 64 * 2);
  float*    dist   = (float*)alloc((size_t)E * 4);
  float*    pooled = (float*)alloc((size_t)NGRAPH * 96 * 4);

  const size_t KSTEP_HALVES = 6 * 32 * 16;  // 3072 halves per k-step
  _Float16* pwE1 = (_Float16*)alloc(2 * KSTEP_HALVES * 2);
  _Float16* pwE2 = (_Float16*)alloc(3 * KSTEP_HALVES * 2);
  _Float16 *pwM1a[NLAYER], *pwM1b[NLAYER], *pwM2[NLAYER];
  _Float16 *pwU1a[NLAYER], *pwU1b[NLAYER], *pwU2[NLAYER];
  for (int l = 0; l < NLAYER; ++l) {
    pwM1a[l] = (_Float16*)alloc(3 * KSTEP_HALVES * 2);
    pwM1b[l] = (_Float16*)alloc(3 * KSTEP_HALVES * 2);
    pwM2[l]  = (_Float16*)alloc(3 * KSTEP_HALVES * 2);
    pwU1a[l] = (_Float16*)alloc(3 * KSTEP_HALVES * 2);
    pwU1b[l] = (_Float16*)alloc(3 * KSTEP_HALVES * 2);
    pwU2[l]  = (_Float16*)alloc(3 * KSTEP_HALVES * 2);
  }
  _Float16* pwP1 = (_Float16*)alloc(3 * KSTEP_HALVES * 2);
  _Float16* pwP2 = (_Float16*)alloc(3 * KSTEP_HALVES * 2);

  auto packW = [&](const float* src, _Float16* dst, int Kact, int ksteps) {
    int total = ksteps * 6 * 32 * 16;
    pack_w_kernel<<<(total + 255) / 256, 256, 0, stream>>>(src, dst, Kact, ksteps);
  };

  // --- pack all weights into WMMA B-fragment layout ---
  packW(embed_w1, pwE1, 35, 2);
  packW(embed_w2, pwE2, 96, 3);
  for (int l = 0; l < NLAYER; ++l) {
    const float* m1 = msg_w1 + (size_t)l * 193 * 96;
    packW(m1,            pwM1a[l], 96, 3);   // rows 0..95   (h[send] part)
    packW(m1 + 96 * 96,  pwM1b[l], 96, 3);   // rows 96..191 (h[rec] part)
    packW(msg_w2 + (size_t)l * 96 * 96, pwM2[l], 96, 3);
    const float* u1 = upd_w1 + (size_t)l * 192 * 96;
    packW(u1,            pwU1a[l], 96, 3);   // rows 0..95   (h part)
    packW(u1 + 96 * 96,  pwU1b[l], 96, 3);   // rows 96..191 (aggr part)
    packW(upd_w2 + (size_t)l * 96 * 96, pwU2[l], 96, 3);
  }
  packW(pre_w1, pwP1, 96, 3);
  packW(pre_w2, pwP2, 96, 3);

  // --- inputs ---
  build_hin_kernel<<<(N * 64 + 255) / 256, 256, 0, stream>>>(x, pe, hin16, N);
  dist_kernel<<<(E + 255) / 256, 256, 0, stream>>>(pos, ei, dist, E);

  const int nodeBlocks = (N + 63) / 64;
  const int edgeBlocks = (E + 63) / 64;

  // --- embed: h = silu(hin·W1+b1)·W2+b2 ---
  fused_mlp2_kernel<<<nodeBlocks, 128, 0, stream>>>(
      hin16, 64, 2, nullptr, nullptr, nullptr,
      pwE1, nullptr, embed_b1, nullptr, nullptr,
      pwE2, embed_b2, /*silu2=*/0, nullptr,
      hA, h16A, nullptr, nullptr, N);

  // --- message-passing layers ---
  for (int l = 0; l < NLAYER; ++l) {
    float*    hcur  = (l & 1) ? hB : hA;
    _Float16* h16c  = (l & 1) ? h16B : h16A;
    float*    hnxt  = (l & 1) ? hA : hB;
    _Float16* h16n  = (l & 1) ? h16A : h16B;

    hipMemsetAsync(aggr, 0, (size_t)N * 96 * 4, stream);

    // edges: msg = silu(silu(h[s]·W1a + h[r]·W1b + dist·w192 + b1)·W2 + b2)
    //        aggr[rec] += msg   (hardware f32 atomics = segment_sum)
    fused_mlp2_kernel<<<edgeBlocks, 128, 0, stream>>>(
        h16c, 96, 3, send, h16c, rec,
        pwM1a[l], pwM1b[l], msg_b1 + (size_t)l * 96,
        dist, msg_w1 + (size_t)l * 193 * 96 + (size_t)192 * 96,
        pwM2[l], msg_b2 + (size_t)l * 96, /*silu2=*/1, nullptr,
        nullptr, nullptr, aggr, rec, E);

    f32_to_f16_kernel<<<(N * 96 + 255) / 256, 256, 0, stream>>>(aggr, aggr16, N * 96);

    // nodes: h' = h + silu(h·U1a + aggr·U1b + b1)·U2 + b2
    fused_mlp2_kernel<<<nodeBlocks, 128, 0, stream>>>(
        h16c, 96, 3, nullptr, aggr16, nullptr,
        pwU1a[l], pwU1b[l], upd_b1 + (size_t)l * 96, nullptr, nullptr,
        pwU2[l], upd_b2 + (size_t)l * 96, /*silu2=*/0, hcur,
        hnxt, h16n, nullptr, nullptr, N);
  }
  // after 4 layers final h is back in A buffers

  // --- pre-MLP + graph pooling (atomic scatter by batch) ---
  hipMemsetAsync(pooled, 0, (size_t)NGRAPH * 96 * 4, stream);
  fused_mlp2_kernel<<<nodeBlocks, 128, 0, stream>>>(
      h16A, 96, 3, nullptr, nullptr, nullptr,
      pwP1, nullptr, pre_b1, nullptr, nullptr,
      pwP2, pre_b2, /*silu2=*/0, nullptr,
      nullptr, nullptr, pooled, batch, N);

  // --- readout ---
  readout_kernel<<<NGRAPH, 96, 0, stream>>>(pooled, ro_w1, ro_b1, ro_w2, ro_b2,
                                            (float*)d_out);
  (void)in_sizes; (void)n_in; (void)out_size; (void)ws_size;
}
